// TemporalGCN_23759759081529
// MI455X (gfx1250) — matrix operand order repaired and checked by reading, not compiled, not run on hardware
//
#include <hip/hip_runtime.h>
#include <math.h>
#include <stdint.h>

#define N_NODES 65536
#define N_EDGES 1048576
#define DIM     128
#define T_STEPS 1024
#define BATCH   64
#define GDIM    512   // 4*OUT_DIM gates

typedef __attribute__((ext_vector_type(16))) _Float16 v16h;
typedef __attribute__((ext_vector_type(8)))  float    v8f;

// ---------------- degree / norm ----------------

__global__ void deg_init_k(float* deg) {
    int i = blockIdx.x * blockDim.x + threadIdx.x;
    if (i < N_NODES) deg[i] = 1.0f;   // self-loop
}

__global__ void deg_scatter_k(const long long* __restrict__ ei, float* deg) {
    int e = blockIdx.x * blockDim.x + threadIdx.x;
    if (e < N_EDGES) {
        int dst = (int)ei[N_EDGES + e];
        atomicAdd(&deg[dst], 1.0f);   // exact integer-valued fp adds: deterministic
    }
}

__global__ void rsqrt_k(float* deg) {
    int i = blockIdx.x * blockDim.x + threadIdx.x;
    if (i < N_NODES) deg[i] = rsqrtf(deg[i]);   // in place -> dinv
}

__global__ void zero_k(float* p, long long n) {
    long long i = (long long)blockIdx.x * blockDim.x + threadIdx.x;
    long long stride = (long long)gridDim.x * blockDim.x;
    for (; i < n; i += stride) p[i] = 0.0f;
}

// ---------------- node GEMM:  out = X @ W   (N_NODES x 128) @ (128 x 128) ----------------
// grid.x = N_NODES/16 row tiles; 256 threads = 8 waves; wave w -> column tile w.
// A-tile (16 rows x 128 cols f32 = 8 KB, contiguous in memory) staged via CDNA5
// GLOBAL_LOAD_ASYNC_TO_LDS_B128 (ASYNCcnt path), no VGPR round-trip.

__global__ __launch_bounds__(256) void gemm_node_k(const float* __restrict__ X,
                                                   const float* __restrict__ W,
                                                   float* __restrict__ out) {
    __shared__ float tileA[16 * DIM];           // 8 KB, at LDS offset of tileA
    const int rowTile = blockIdx.x;
    const int lane = threadIdx.x & 31;
    const int wave = threadIdx.x >> 5;          // 0..7 -> N tile
    const int l15  = lane & 15;
    const int halfS = lane >> 4;
    const int base = rowTile * 16;

    // --- async stage: 512 chunks of 16 B; each wave-wide op moves 32 x 16 B ---
    {
        // low 32 bits of the flat shared address == wave-relative LDS byte offset
        const unsigned ldsBase = (unsigned)(uintptr_t)tileA;
        const unsigned long long gBase =
            (unsigned long long)(uintptr_t)X + (unsigned long long)base * (DIM * 4ull);
        for (int j = threadIdx.x; j < 512; j += 256) {
            unsigned loff = ldsBase + (unsigned)(j * 16);
            unsigned long long g = gBase + (unsigned long long)j * 16ull;
            asm volatile("global_load_async_to_lds_b128 %0, %1, off"
                         :: "v"(loff), "v"(g) : "memory");
        }
        asm volatile("s_wait_asynccnt 0x0" ::: "memory");
    }
    __syncthreads();

    const int colB = wave * 16 + l15;           // B / C column
    v8f acc = {};
#pragma unroll
    for (int kc = 0; kc < 4; ++kc) {
        v16h a, b;
#pragma unroll
        for (int e = 0; e < 16; ++e) {
            int ka = kc * 32 + ((e < 8) ? e : e + 8) + 8 * halfS;   // A lane layout
            a[e] = (_Float16)tileA[l15 * DIM + ka];
            int kb = kc * 32 + ((e < 8) ? e : e + 8) + 8 * halfS;   // B lane layout
            b[e] = (_Float16)W[kb * DIM + colB];
        }
        acc = __builtin_amdgcn_wmma_f32_16x16x32_f16(false, a, false, b,
                                                     (short)0, acc, false, false);
    }
    const int mBase = base + 8 * halfS;
#pragma unroll
    for (int r = 0; r < 8; ++r)
        out[(mBase + r) * DIM + colB] = acc[r];
}

// ---------------- edge aggregation: agg[dst] += hw[src] * dinv[src]*dinv[dst] ----------------
// one block per edge, 128 lanes = one feature each: fully coalesced row move, L2-resident.

__global__ __launch_bounds__(128) void edge_agg_k(const long long* __restrict__ ei,
                                                  const float* __restrict__ hw,
                                                  const float* __restrict__ dinv,
                                                  float* __restrict__ agg) {
    const int e   = blockIdx.x;
    const int src = (int)ei[e];
    const int dst = (int)ei[N_EDGES + e];
    const float norm = dinv[src] * dinv[dst];
    const int c = threadIdx.x;
    atomicAdd(&agg[dst * DIM + c], hw[src * DIM + c] * norm);
}

// ---------------- finalize:  h = relu(agg + hw*dinv^2 + b) ----------------

__global__ __launch_bounds__(128) void finalize_relu_k(const float* __restrict__ hw,
                                                       const float* __restrict__ agg,
                                                       const float* __restrict__ dinv,
                                                       const float* __restrict__ bias,
                                                       float* __restrict__ out) {
    const int n = blockIdx.x;
    const int c = threadIdx.x;
    const float d2 = dinv[n] * dinv[n];
    const float v = agg[n * DIM + c] + hw[n * DIM + c] * d2 + bias[c];
    out[n * DIM + c] = v > 0.0f ? v : 0.0f;
}

// ---------------- persistent LSTM: 1 workgroup, 32 waves, weights in VGPRs ----------------
// wave w owns gate-column tile w (16 of 512 cols); B-fragments of W_ih/W_hh stay in registers
// for all 1024 steps. h, c, staged x_t, and the 64x512 gate matrix live in LDS (224 KB dyn).
// Next timestep's strided x_t gather is prefetched (global_prefetch_b8) to hide latency in
// the strictly sequential recurrence.

__global__ __launch_bounds__(1024) void lstm_k(const float* __restrict__ h2,
                                               const float* __restrict__ W_ih,
                                               const float* __restrict__ W_hh,
                                               const float* __restrict__ b_ih,
                                               const float* __restrict__ b_hh,
                                               float* __restrict__ out) {
    extern __shared__ float smem[];
    float* sh = smem;                       // [64*128] h state
    float* sc = smem + BATCH * DIM;         // [64*128] c state
    float* sx = smem + 2 * BATCH * DIM;     // [64*128] staged x_t
    float* sg = smem + 3 * BATCH * DIM;     // [64*512] gates

    const int tid  = threadIdx.x;
    const int lane = tid & 31;
    const int wave = tid >> 5;              // 0..31 -> N tile of gates
    const int l15  = lane & 15;
    const int halfS = lane >> 4;
    const int ncol = wave * 16 + l15;       // gate column 0..511

    for (int i = tid; i < BATCH * DIM; i += 1024) { sh[i] = 0.0f; sc[i] = 0.0f; }

    // persistent B fragments: B[k][n] = W[n][k]  (gates = x @ W^T)
    v16h Bih[4], Bhh[4];
#pragma unroll
    for (int kc = 0; kc < 4; ++kc) {
#pragma unroll
        for (int e = 0; e < 16; ++e) {
            int k = kc * 32 + ((e < 8) ? e : e + 8) + 8 * halfS;
            Bih[kc][e] = (_Float16)W_ih[ncol * DIM + k];
            Bhh[kc][e] = (_Float16)W_hh[ncol * DIM + k];
        }
    }
    const float bias_n = b_ih[ncol] + b_hh[ncol];
    __syncthreads();

    for (int t = 0; t < T_STEPS; ++t) {
        // stage x_t = h2[b*T + t, :]; prefetch t+1's rows while we're at it
        for (int i = tid; i < BATCH * DIM; i += 1024) {
            int b = i >> 7, c = i & 127;
            sx[i] = h2[(b * T_STEPS + t) * DIM + c];
            if (t + 1 < T_STEPS)
                __builtin_prefetch(&h2[(b * T_STEPS + t + 1) * DIM + c], 0, 0);
        }
        __syncthreads();

        // gates[64 x 512] = bias + x_t @ W_ih^T + h @ W_hh^T  (4 M-tiles per wave)
#pragma unroll
        for (int mt = 0; mt < 4; ++mt) {
            v8f acc;
#pragma unroll
            for (int r = 0; r < 8; ++r) acc[r] = bias_n;
            const int mrow = mt * 16 + l15;
#pragma unroll
            for (int kc = 0; kc < 4; ++kc) {
                v16h ax, ah;
#pragma unroll
                for (int e = 0; e < 16; ++e) {
                    int k = kc * 32 + ((e < 8) ? e : e + 8) + 8 * halfS;
                    ax[e] = (_Float16)sx[mrow * DIM + k];
                    ah[e] = (_Float16)sh[mrow * DIM + k];
                }
                acc = __builtin_amdgcn_wmma_f32_16x16x32_f16(false, ax, false, Bih[kc],
                                                             (short)0, acc, false, false);
                acc = __builtin_amdgcn_wmma_f32_16x16x32_f16(false, ah, false, Bhh[kc],
                                                             (short)0, acc, false, false);
            }
#pragma unroll
            for (int r = 0; r < 8; ++r) {
                int m = mt * 16 + r + 8 * halfS;
                sg[m * GDIM + ncol] = acc[r];
            }
        }
        __syncthreads();

        // elementwise LSTM cell (torch gate order i,f,g,o)
        for (int i = tid; i < BATCH * DIM; i += 1024) {
            int b = i >> 7, c = i & 127;
            float gi = sg[b * GDIM + c];
            float gf = sg[b * GDIM + 128 + c];
            float gg = sg[b * GDIM + 256 + c];
            float go = sg[b * GDIM + 384 + c];
            gi = 1.0f / (1.0f + expf(-gi));
            gf = 1.0f / (1.0f + expf(-gf));
            gg = tanhf(gg);
            go = 1.0f / (1.0f + expf(-go));
            float cn = gf * sc[i] + gi * gg;
            sc[i] = cn;
            sh[i] = go * tanhf(cn);
        }
        __syncthreads();
    }

    for (int i = tid; i < BATCH * DIM; i += 1024) out[i] = sh[i];
}

// ---------------- launch ----------------

extern "C" void kernel_launch(void* const* d_in, const int* in_sizes, int n_in,
                              void* d_out, int out_size, void* d_ws, size_t ws_size,
                              hipStream_t stream) {
    const float*      x    = (const float*)d_in[0];
    const long long*  ei   = (const long long*)d_in[1];   // int64 [2, E]
    // d_in[2] = batch_size (unused; B=64 baked in)
    const float*      W1   = (const float*)d_in[3];
    const float*      b1   = (const float*)d_in[4];
    const float*      W2   = (const float*)d_in[5];
    const float*      b2   = (const float*)d_in[6];
    const float*      W_ih = (const float*)d_in[7];
    const float*      W_hh = (const float*)d_in[8];
    const float*      b_ih = (const float*)d_in[9];
    const float*      b_hh = (const float*)d_in[10];
    float* out = (float*)d_out;

    // workspace: dinv (1 MB slot) + three 32 MB node-feature buffers (~97 MB total)
    char*  ws   = (char*)d_ws;
    float* dinv = (float*)ws;
    float* hw   = (float*)(ws + (1ll << 20));
    float* agg  = (float*)(ws + (1ll << 20) + (32ll << 20));
    float* h    = (float*)(ws + (1ll << 20) + (64ll << 20));
    const long long featN = (long long)N_NODES * DIM;

    // degree -> dinv
    deg_init_k   <<<N_NODES / 256, 256, 0, stream>>>(dinv);
    deg_scatter_k<<<N_EDGES / 256, 256, 0, stream>>>(ei, dinv);
    rsqrt_k      <<<N_NODES / 256, 256, 0, stream>>>(dinv);

    // GCN layer 1
    zero_k        <<<4096, 256, 0, stream>>>(agg, featN);
    gemm_node_k   <<<N_NODES / 16, 256, 0, stream>>>(x, W1, hw);
    edge_agg_k    <<<N_EDGES, 128, 0, stream>>>(ei, hw, dinv, agg);
    finalize_relu_k<<<N_NODES, 128, 0, stream>>>(hw, agg, dinv, b1, h);

    // GCN layer 2
    zero_k        <<<4096, 256, 0, stream>>>(agg, featN);
    gemm_node_k   <<<N_NODES / 16, 256, 0, stream>>>(h, W2, hw);
    edge_agg_k    <<<N_EDGES, 128, 0, stream>>>(ei, hw, dinv, agg);
    finalize_relu_k<<<N_NODES, 128, 0, stream>>>(hw, agg, dinv, b2, h);

    // persistent LSTM: dynamic LDS = (3*64*128 + 64*512) floats = 224 KB (< 320 KB/WGP)
    const size_t lds_bytes = (size_t)(3 * BATCH * DIM + BATCH * GDIM) * sizeof(float);
    lstm_k<<<1, 1024, lds_bytes, stream>>>(h, W_ih, W_hh, b_ih, b_hh, out);
}
